// TFConvBertSelfAttention_55336358642168
// MI455X (gfx1250) — matrix operand order repaired and checked by reading, not compiled, not run on hardware
//
#include <hip/hip_runtime.h>

// ---------------------------------------------------------------- constants
#define C_INDIM 768
#define AHS     384
#define NHEAD   6
#define HDIM    64
#define KW      9
#define HK      54      // NHEAD*KW
#define BATCH   4
#define SEQ     2048
#define ROWS    (BATCH * SEQ)   // 8192
#define NCAT    (4 * AHS)       // fused q|k|v|co projection width = 1536

typedef __attribute__((ext_vector_type(16))) __bf16 v16bf;
typedef __attribute__((ext_vector_type(8)))  float  v8f;
typedef __attribute__((ext_vector_type(4)))  int    v4i_t;

// pointer-to-AS1-v4i and pointer-to-AS3-v4i (pointee is address-space qualified)
typedef __attribute__((address_space(1))) v4i_t* global_v4i_ptr;
typedef __attribute__((address_space(3))) v4i_t* lds_v4i_ptr;

// ---------------------------------------------------------------- async global->LDS (gfx1250), guarded
#if defined(__has_builtin)
#  if __has_builtin(__builtin_amdgcn_global_load_async_to_lds_b128)
#    define HAVE_ASYNC_B128 1
#  endif
#  if __has_builtin(__builtin_amdgcn_s_wait_asynccnt)
#    define HAVE_WAIT_ASYNC 1
#  endif
#endif

__device__ __forceinline__ void async_copy_b128(const void* g, void* l) {
#if defined(HAVE_ASYNC_B128)
    __builtin_amdgcn_global_load_async_to_lds_b128(
        (global_v4i_ptr)g, (lds_v4i_ptr)l, 0, 0);
#else
    *(uint4*)l = *(const uint4*)g;
#endif
}

#if defined(HAVE_ASYNC_B128)
#  if defined(HAVE_WAIT_ASYNC)
#    define ASYNC_WAIT(n) __builtin_amdgcn_s_wait_asynccnt(n)
#  else
#    define ASYNC_WAIT(n) asm volatile("s_wait_asynccnt %0" :: "i"(n) : "memory")
#  endif
#else
#  define ASYNC_WAIT(n) do { } while (0)
#endif

// ---------------------------------------------------------------- WMMA core
__device__ __forceinline__ v8f wmma_bf16(v16bf a, v16bf b, v8f c) {
    // D = A(16x32 bf16) * B(32x16 bf16) + C(16x16 f32)
    return __builtin_amdgcn_wmma_f32_16x16x32_bf16(
        false, a, false, b, (short)0, c, false, false);
}

// A fragment (16x32, bf16) from LDS row-major [row][ld]:
// lanes 0-15: M=lane, K = 0..15
// lanes16-31: M=lane-16, K = 16..31
__device__ __forceinline__ v16bf load_a_frag(const __bf16* lds, int row_base,
                                             int ld, int kbase) {
    int lane = threadIdx.x & 31;
    int m = lane & 15;
    int kh = (lane >> 4) ? 16 : 0;
    const __bf16* p = lds + (row_base + m) * ld + kbase + kh;
    v16bf a;
#pragma unroll
    for (int i = 0; i < 16; ++i) a[i] = p[i];
    return a;
}

// B fragment (32x16, bf16) from LDS stored transposed: Bt[n][ld] (contiguous k)
// lanes 0-15: N=lane, K = 0..15 ; lanes 16-31: N=lane-16, K = 16..31
__device__ __forceinline__ v16bf load_bT_frag(const __bf16* lds, int col_base,
                                              int ld, int kbase) {
    int lane = threadIdx.x & 31;
    int n = lane & 15;
    int ko = (lane >> 4) ? 16 : 0;
    const __bf16* p = lds + (col_base + n) * ld + kbase + ko;
    v16bf b;
#pragma unroll
    for (int i = 0; i < 16; ++i) b[i] = p[i];
    return b;
}

// ---------------------------------------------------------------- small prep kernels
__global__ __launch_bounds__(256)
void cvt_f32_bf16_kernel(const float* __restrict__ in, __bf16* __restrict__ out, int n) {
    for (int i = blockIdx.x * 256 + threadIdx.x; i < n; i += gridDim.x * 256)
        out[i] = (__bf16)in[i];
}

// W[K][Nseg] f32 -> Wt[n][K] bf16 (transposed, coalesced writes)
__global__ __launch_bounds__(256)
void cvt_transpose_kernel(const float* __restrict__ W, __bf16* __restrict__ Wt,
                          int Kd, int Nseg) {
    int i = blockIdx.x * 256 + threadIdx.x;
    if (i >= Nseg * Kd) return;
    int n = i / Kd, k = i % Kd;
    Wt[i] = (__bf16)W[(size_t)k * Nseg + n];
}

__global__ __launch_bounds__(256)
void copy_f32_kernel(const float* __restrict__ in, float* __restrict__ out, int n) {
    int i = blockIdx.x * 256 + threadIdx.x;
    if (i < n) out[i] = in[i];
}

// ---------------------------------------------------------------- depthwise conv (SAME, K=9)
__global__ __launch_bounds__(256)
void depthwise_kernel(const float* __restrict__ hs, const float* __restrict__ dwk,
                      __bf16* __restrict__ out) {
    int i = blockIdx.x * 256 + threadIdx.x;
    if (i >= ROWS * C_INDIM) return;
    int c = i % C_INDIM;
    int row = i / C_INDIM;
    int s = row % SEQ, b = row / SEQ;
    float acc = 0.f;
#pragma unroll
    for (int t = 0; t < KW; ++t) {
        int ss = s + t - (KW - 1) / 2;
        if (ss >= 0 && ss < SEQ)
            acc += hs[((size_t)b * SEQ + ss) * C_INDIM + c] * dwk[t * C_INDIM + c];
    }
    out[i] = (__bf16)acc;
}

// ---------------------------------------------------------------- WMMA GEMM (double-buffered, async fill)
// C[seg][M,Nout](bf16) = A[M,K] @ Bt^T + bias   where Bt is [N][K] (pre-transposed),
// N = total fused width, output split into N/Nout consecutive [M,Nout] buffers.
__global__ __launch_bounds__(256)
void gemm_bf16_kernel(const __bf16* __restrict__ A, const __bf16* __restrict__ BtW,
                      const float* __restrict__ bias, __bf16* __restrict__ Cout,
                      int Kd, int Nout) {
    __shared__ __align__(16) __bf16 As[2][128 * 32];
    __shared__ __align__(16) __bf16 Bs[2][64 * 32];
    int tid = threadIdx.x, wave = tid >> 5, lane = tid & 31;
    int m0 = blockIdx.y * 128, n0 = blockIdx.x * 64;

    // per-thread load slots (contiguous 16B chunks)
    int arow = tid >> 1, akc = (tid & 1) * 16;       // A: 128x32, 32B/thread
    int brow = tid >> 2, bkc = (tid & 3) * 8;        // B: 64x32, 16B/thread
    const __bf16* gA = A + (size_t)(m0 + arow) * Kd + akc;
    const __bf16* gB = BtW + (size_t)(n0 + brow) * Kd + bkc;

    int nt = Kd / 32;
    async_copy_b128(gA,     &As[0][arow * 32 + akc]);
    async_copy_b128(gA + 8, &As[0][arow * 32 + akc + 8]);
    async_copy_b128(gB,     &Bs[0][brow * 32 + bkc]);

    v8f acc[4] = {};
    for (int i = 0; i < nt; ++i) {
        int cur = i & 1, nxt = cur ^ 1;
        if (i + 1 < nt) {
            const __bf16* nA = gA + (size_t)(i + 1) * 32;
            const __bf16* nB = gB + (size_t)(i + 1) * 32;
            async_copy_b128(nA,     &As[nxt][arow * 32 + akc]);
            async_copy_b128(nA + 8, &As[nxt][arow * 32 + akc + 8]);
            async_copy_b128(nB,     &Bs[nxt][brow * 32 + bkc]);
            ASYNC_WAIT(3);        // current tile's 3 async ops complete; next 3 in flight
        } else {
            ASYNC_WAIT(0);
        }
        __syncthreads();
        v16bf af = load_a_frag(As[cur], wave * 16, 32, 0);
#pragma unroll
        for (int c = 0; c < 4; ++c) {
            v16bf bf = load_bT_frag(Bs[cur], c * 16, 32, 0);
            acc[c] = wmma_bf16(af, bf, acc[c]);
        }
        __syncthreads();          // protect buffer before next-next overwrite
    }

    int seg = n0 / Nout;
    int ncol0 = n0 % Nout;
    int n = lane & 15, mb = (lane >> 4) * 8;
    __bf16* Cseg = Cout + (size_t)seg * ROWS * Nout;
#pragma unroll
    for (int c = 0; c < 4; ++c) {
        int col = ncol0 + c * 16 + n;
        float bv = bias[n0 + c * 16 + n];
#pragma unroll
        for (int r = 0; r < 8; ++r) {
            int row = m0 + wave * 16 + mb + r;
            Cseg[(size_t)row * Nout + col] = (__bf16)(acc[c][r] + bv);
        }
    }
}

// ---------------------------------------------------------------- span kernel predictor: ck = (q*key_conv) @ Wck + bck
__global__ __launch_bounds__(256)
void ck_kernel(const __bf16* __restrict__ qb, const __bf16* __restrict__ kcb,
               const float* __restrict__ Wck, const float* __restrict__ bck,
               float* __restrict__ ck) {
    int i = blockIdx.x * 256 + threadIdx.x;
    if (i >= ROWS * HK) return;
    int o = i % HK, row = i / HK;
    const __bf16* qr = qb + (size_t)row * AHS;
    const __bf16* kr = kcb + (size_t)row * AHS;
    float acc = bck[o];
    for (int j = 0; j < AHS; ++j)
        acc += (float)qr[j] * (float)kr[j] * Wck[j * HK + o];
    ck[i] = acc;
}

__global__ __launch_bounds__(256)
void ck_softmax_kernel(float* __restrict__ ck) {
    int i = blockIdx.x * 256 + threadIdx.x;
    if (i >= ROWS * NHEAD) return;
    float* p = ck + (size_t)i * KW;
    float m = p[0];
#pragma unroll
    for (int t = 1; t < KW; ++t) m = fmaxf(m, p[t]);
    float s = 0.f;
#pragma unroll
    for (int t = 0; t < KW; ++t) { float e = __expf(p[t] - m); p[t] = e; s += e; }
    float inv = 1.f / s;
#pragma unroll
    for (int t = 0; t < KW; ++t) p[t] *= inv;
}

// dynamic conv mix: out[:, 384:768] = einsum('bshdk,bshk->bshd')
__global__ __launch_bounds__(256)
void conv_out_kernel(const __bf16* __restrict__ co, const float* __restrict__ ck,
                     float* __restrict__ out) {
    int i = blockIdx.x * 256 + threadIdx.x;
    if (i >= ROWS * AHS) return;
    int d = i % HDIM;
    int t2 = i / HDIM;
    int h = t2 % NHEAD, row = t2 / NHEAD;
    int s = row % SEQ, b = row / SEQ;
    const float* w = ck + ((size_t)row * NHEAD + h) * KW;
    float acc = 0.f;
#pragma unroll
    for (int t = 0; t < KW; ++t) {
        int ss = s + t - (KW - 1) / 2;
        if (ss >= 0 && ss < SEQ)
            acc += (float)co[((size_t)b * SEQ + ss) * AHS + h * HDIM + d] * w[t];
    }
    out[(size_t)row * (2 * AHS) + AHS + h * HDIM + d] = acc;
}

// ---------------------------------------------------------------- flash attention (WMMA QK^T and PV, online softmax)
__global__ __launch_bounds__(256)
void attention_kernel(const __bf16* __restrict__ qb, const __bf16* __restrict__ kb,
                      const __bf16* __restrict__ vb, const float* __restrict__ amask,
                      const float* __restrict__ hmask, float* __restrict__ out) {
    __shared__ __align__(16) __bf16 Qs[128 * 64];   // q rows x d
    __shared__ __align__(16) __bf16 Ks[64 * 64];    // key x d (B-frag: contiguous d)
    __shared__ __align__(16) __bf16 Vt[64 * 64];    // d x key (B-frag: contiguous key)
    __shared__ __align__(16) __bf16 Ps[8][16 * 64]; // per-wave P staging

    int tid = threadIdx.x, wave = tid >> 5, lane = tid & 31;
    int b = blockIdx.z, h = blockIdx.y, q0 = blockIdx.x * 128;
    int n = lane & 15, mb = (lane >> 4) * 8;

    {   // Q tile async: 128x64, 64B/thread
        int r = tid >> 1, d0 = (tid & 1) * 32;
        const __bf16* g = qb + ((size_t)(b * SEQ + q0 + r)) * AHS + h * HDIM + d0;
#pragma unroll
        for (int j = 0; j < 4; ++j)
            async_copy_b128(g + j * 8, &Qs[r * 64 + d0 + j * 8]);
    }
    ASYNC_WAIT(0);
    __syncthreads();
    v16bf qa0 = load_a_frag(Qs, wave * 16, 64, 0);
    v16bf qa1 = load_a_frag(Qs, wave * 16, 64, 32);

    v8f oacc[4] = {};
    float row_m[8], row_l[8];
#pragma unroll
    for (int r = 0; r < 8; ++r) { row_m[r] = -1e30f; row_l[r] = 0.f; }

    for (int kv0 = 0; kv0 < SEQ; kv0 += 64) {
        __syncthreads();
        {   // K tile async: 64x64, 32B/thread
            int r = tid >> 2, d0 = (tid & 3) * 16;
            const __bf16* g = kb + ((size_t)(b * SEQ + kv0 + r)) * AHS + h * HDIM + d0;
            async_copy_b128(g,     &Ks[r * 64 + d0]);
            async_copy_b128(g + 8, &Ks[r * 64 + d0 + 8]);
        }
        // V tile transposed on store (async can't transpose)
        for (int idx = tid; idx < 64 * 64; idx += 256) {
            int r = idx >> 6, d = idx & 63;
            Vt[d * 64 + r] = vb[((size_t)(b * SEQ + kv0 + r)) * AHS + h * HDIM + d];
        }
        ASYNC_WAIT(0);
        __syncthreads();

        // S = Q K^T * 1/sqrt(D) + mask
        v8f sc[4] = {};
#pragma unroll
        for (int c = 0; c < 4; ++c) {
            v16bf b0 = load_bT_frag(Ks, c * 16, 64, 0);
            sc[c] = wmma_bf16(qa0, b0, sc[c]);
            v16bf b1 = load_bT_frag(Ks, c * 16, 64, 32);
            sc[c] = wmma_bf16(qa1, b1, sc[c]);
        }
#pragma unroll
        for (int c = 0; c < 4; ++c) {
            float mv = amask[(size_t)b * SEQ + kv0 + c * 16 + n];
#pragma unroll
            for (int r = 0; r < 8; ++r) sc[c][r] = sc[c][r] * 0.125f + mv;
        }

        // row max across 16-lane half (rows m / m+8 live in the two halves)
        float tmax[8];
#pragma unroll
        for (int r = 0; r < 8; ++r) {
            float m = sc[0][r];
#pragma unroll
            for (int c = 1; c < 4; ++c) m = fmaxf(m, sc[c][r]);
            m = fmaxf(m, __shfl_xor(m, 1, 32));
            m = fmaxf(m, __shfl_xor(m, 2, 32));
            m = fmaxf(m, __shfl_xor(m, 4, 32));
            m = fmaxf(m, __shfl_xor(m, 8, 32));
            tmax[r] = m;
        }
        float corr[8];
#pragma unroll
        for (int r = 0; r < 8; ++r) {
            float mnew = fmaxf(row_m[r], tmax[r]);
            corr[r] = __expf(row_m[r] - mnew);
            row_m[r] = mnew;
        }
        float rsum[8];
#pragma unroll
        for (int r = 0; r < 8; ++r) rsum[r] = 0.f;
#pragma unroll
        for (int c = 0; c < 4; ++c)
#pragma unroll
            for (int r = 0; r < 8; ++r) {
                float p = __expf(sc[c][r] - row_m[r]);
                rsum[r] += p;
                Ps[wave][(mb + r) * 64 + c * 16 + n] = (__bf16)p;
            }
#pragma unroll
        for (int r = 0; r < 8; ++r) {
            float s = rsum[r];
            s += __shfl_xor(s, 1, 32);
            s += __shfl_xor(s, 2, 32);
            s += __shfl_xor(s, 4, 32);
            s += __shfl_xor(s, 8, 32);
            row_l[r] = row_l[r] * corr[r] + s;
        }
#pragma unroll
        for (int c = 0; c < 4; ++c)
#pragma unroll
            for (int r = 0; r < 8; ++r) oacc[c][r] *= corr[r];

        // per-wave P stores -> A-layout reads (DS in-order within wave; be explicit)
        asm volatile("s_wait_dscnt 0" ::: "memory");

#pragma unroll
        for (int ks = 0; ks < 2; ++ks) {
            v16bf pa = load_a_frag(&Ps[wave][0], 0, 64, ks * 32);
#pragma unroll
            for (int c = 0; c < 4; ++c) {
                v16bf bv = load_bT_frag(Vt, c * 16, 64, ks * 32);
                oacc[c] = wmma_bf16(pa, bv, oacc[c]);
            }
        }
    }

    float hm = hmask[h];
#pragma unroll
    for (int c = 0; c < 4; ++c)
#pragma unroll
        for (int r = 0; r < 8; ++r) {
            int row = q0 + wave * 16 + mb + r;
            out[((size_t)(b * SEQ + row)) * (2 * AHS) + h * HDIM + c * 16 + n] =
                oacc[c][r] * hm / row_l[r];
        }
}

// ---------------------------------------------------------------- launch
extern "C" void kernel_launch(void* const* d_in, const int* in_sizes, int n_in,
                              void* d_out, int out_size, void* d_ws, size_t ws_size,
                              hipStream_t stream) {
    const float* hs    = (const float*)d_in[0];
    const float* amask = (const float*)d_in[1];
    const float* hmask = (const float*)d_in[2];
    const float* Wq = (const float*)d_in[3];  const float* bq = (const float*)d_in[4];
    const float* Wk = (const float*)d_in[5];  const float* bk = (const float*)d_in[6];
    const float* Wv = (const float*)d_in[7];  const float* bv = (const float*)d_in[8];
    const float* dwk = (const float*)d_in[9];
    const float* pw  = (const float*)d_in[10]; const float* cbias = (const float*)d_in[11];
    const float* Wck = (const float*)d_in[12]; const float* bck  = (const float*)d_in[13];
    const float* Wco = (const float*)d_in[14]; const float* bco  = (const float*)d_in[15];
    float* out = (float*)d_out;

    char* ws = (char*)d_ws;
    size_t off = 0;
    auto alloc = [&](size_t bytes) -> void* {
        void* p = ws + off;
        off = (off + bytes + 255) & ~(size_t)255;
        return p;
    };
    __bf16* hbf   = (__bf16*)alloc((size_t)ROWS * C_INDIM * 2);
    __bf16* dwbf  = (__bf16*)alloc((size_t)ROWS * C_INDIM * 2);
    __bf16* projb = (__bf16*)alloc((size_t)4 * ROWS * AHS * 2);   // q|k|v|co contiguous
    __bf16* kcbf  = (__bf16*)alloc((size_t)ROWS * AHS * 2);
    __bf16* wcatT = (__bf16*)alloc((size_t)NCAT * C_INDIM * 2);   // [1536][768] transposed
    __bf16* pwT   = (__bf16*)alloc((size_t)AHS * C_INDIM * 2);    // [384][768] transposed
    float*  bcat  = (float*)alloc((size_t)NCAT * 4);
    float*  ck    = (float*)alloc((size_t)ROWS * HK * 4);

    __bf16* qbf  = projb;
    __bf16* kbf  = projb + (size_t)1 * ROWS * AHS;
    __bf16* vbf  = projb + (size_t)2 * ROWS * AHS;
    __bf16* cobf = projb + (size_t)3 * ROWS * AHS;

    const int T = 256;
    const int WKN = C_INDIM * AHS;           // 294912 elements per weight
    const int WB  = (WKN + T - 1) / T;

    // 1) converts + weight transposes + bias concat
    cvt_f32_bf16_kernel<<<8192, T, 0, stream>>>(hs, hbf, ROWS * C_INDIM);
    cvt_transpose_kernel<<<WB, T, 0, stream>>>(Wq,  wcatT + (size_t)0 * WKN, C_INDIM, AHS);
    cvt_transpose_kernel<<<WB, T, 0, stream>>>(Wk,  wcatT + (size_t)1 * WKN, C_INDIM, AHS);
    cvt_transpose_kernel<<<WB, T, 0, stream>>>(Wv,  wcatT + (size_t)2 * WKN, C_INDIM, AHS);
    cvt_transpose_kernel<<<WB, T, 0, stream>>>(Wco, wcatT + (size_t)3 * WKN, C_INDIM, AHS);
    cvt_transpose_kernel<<<WB, T, 0, stream>>>(pw,  pwT, C_INDIM, AHS);
    copy_f32_kernel<<<2, T, 0, stream>>>(bq,  bcat + 0 * AHS, AHS);
    copy_f32_kernel<<<2, T, 0, stream>>>(bk,  bcat + 1 * AHS, AHS);
    copy_f32_kernel<<<2, T, 0, stream>>>(bv,  bcat + 2 * AHS, AHS);
    copy_f32_kernel<<<2, T, 0, stream>>>(bco, bcat + 3 * AHS, AHS);
    // 2) depthwise conv
    depthwise_kernel<<<(ROWS * C_INDIM + T - 1) / T, T, 0, stream>>>(hs, dwk, dwbf);
    // 3) fused projection GEMM: [8192,768] @ [768,1536] -> q|k|v|co  (A-tile loaded once)
    dim3 ggcat(NCAT / 64, ROWS / 128);
    gemm_bf16_kernel<<<ggcat, T, 0, stream>>>(hbf, wcatT, bcat, projb, C_INDIM, AHS);
    // key_conv GEMM: [8192,768] @ [768,384]
    dim3 ggkc(AHS / 64, ROWS / 128);
    gemm_bf16_kernel<<<ggkc, T, 0, stream>>>(dwbf, pwT, cbias, kcbf, C_INDIM, AHS);
    // 4) span kernel + softmax + dynamic conv mix (second half of output)
    ck_kernel<<<(ROWS * HK + T - 1) / T, T, 0, stream>>>(qbf, kcbf, Wck, bck, ck);
    ck_softmax_kernel<<<(ROWS * NHEAD + T - 1) / T, T, 0, stream>>>(ck);
    conv_out_kernel<<<(ROWS * AHS + T - 1) / T, T, 0, stream>>>(cobf, ck, out);
    // 5) flash attention (first half of output)
    dim3 ga(SEQ / 128, NHEAD, BATCH);
    attention_kernel<<<ga, T, 0, stream>>>(qbf, kbf, vbf, amask, hmask, out);
}
